// HeteroPredictor_3917010174735
// MI455X (gfx1250) — matrix operand order repaired and checked by reading, not compiled, not run on hardware
//
#include <hip/hip_runtime.h>

// ---------------------------------------------------------------------------
// HeteroPredictor on MI455X (gfx1250, wave32).
// Edge-message MLPs + node update mapped onto V_WMMA_F32_16X16X32_BF16.
// Round-3 change: edge feature rows are gathered with coalesced float4
// (b128) loads into a *permuted* LDS bf16 tile ([feat | pos | dis | 0-pad]),
// with the matching permutation applied to the W1 B-fragments. Pad columns
// are zeroed once per kernel instead of per tile.
// ---------------------------------------------------------------------------

#define N_S   50000
#define N_A   50000
#define NE    1600000
#define U_DIM 16
#define H_DIM 64
#define MLP_H 32

typedef __attribute__((ext_vector_type(16))) __bf16 v16bf;
typedef __attribute__((ext_vector_type(4)))  __bf16 v4bf;
typedef __attribute__((ext_vector_type(8)))  float  v8f;

#define HID_STRIDE 34   // hidden tile: 32 cols + pad (bf16 units)

// Unpermuted B fragment (used for W2): lanes 0-15 hold K {0..7,16..23},
// lanes 16-31 hold K {8..15,24..31}; N = lane & 15.
static __device__ inline v16bf load_B_frag(const float* __restrict__ W, int Kdim,
                                           int Nfull, int kc, int nblk, int lane) {
    const int n = nblk * 16 + (lane & 15);
    const int kbase = kc * 32 + ((lane & 16) ? 8 : 0);
    v16bf b;
#pragma unroll
    for (int j = 0; j < 16; ++j) {
        const int k = kbase + (j < 8 ? j : j + 8);
        const float v = (k < Kdim) ? W[k * Nfull + n] : 0.0f;
        b[j] = (__bf16)v;
    }
    return b;
}

// Permuted W1 fragment matching the LDS feature layout:
//   p in [0, FDIM)        -> original k = 5 + p        (feat block)
//   p in [FDIM, FDIM+5)   -> original k = p - FDIM     (pos/pos/dis block)
//   p >= FDIM+5           -> zero pad
template <int FDIM>
static __device__ inline v16bf load_B1_perm(const float* __restrict__ W1,
                                            int kc, int nblk, int lane) {
    const int n = nblk * 16 + (lane & 15);
    const int pbase = kc * 32 + ((lane & 16) ? 8 : 0);
    v16bf b;
#pragma unroll
    for (int j = 0; j < 16; ++j) {
        const int p = pbase + (j < 8 ? j : j + 8);
        const int k = (p < FDIM) ? (5 + p) : ((p < FDIM + 5) ? (p - FDIM) : -1);
        const float v = (k >= 0) ? W1[k * MLP_H + n] : 0.0f;
        b[j] = (__bf16)v;
    }
    return b;
}

// A-matrix (16 x 32) bf16 fragment from a row-major LDS tile at chunk kc:
// lanes 0-15: M=lane, K {0..7,16..23}; lanes 16-31: M=lane-16, K {8..15,24..31}.
static __device__ inline v16bf load_A_lds(const __bf16* __restrict__ tile,
                                          int stride, int kc, int lane) {
    const int m  = lane & 15;
    const int kb = kc * 32 + ((lane & 16) ? 8 : 0);
    v16bf a;
#pragma unroll
    for (int j = 0; j < 8; ++j) {
        a[j]     = tile[m * stride + kb + j];
        a[8 + j] = tile[m * stride + kb + 16 + j];
    }
    return a;
}

// ---------------------------------------------------------------------------
// Edge-message kernel: one wave == 16 edges.
//   coalesced float4 gather -> LDS bf16 tile -> WMMA MLP -> tanh ->
//   fp32 atomic scatter-add into outSum[dst].
// ---------------------------------------------------------------------------
template <int FDIM>
__global__ __launch_bounds__(256)
void edge_msg_kernel(const int* __restrict__ srcA, const int* __restrict__ dstA,
                     const float* __restrict__ disA,
                     const float* __restrict__ posSrc, const float* __restrict__ posS,
                     const float* __restrict__ feat,
                     const float* __restrict__ W1, const float* __restrict__ b1,
                     const float* __restrict__ W2, const float* __restrict__ b2,
                     float* __restrict__ outSum) {
    constexpr int KDIM    = 5 + FDIM;           // logical K (21 or 69)
    constexpr int NCHUNK  = (KDIM + 31) / 32;   // 1 or 3
    constexpr int KPAD    = NCHUNK * 32;        // 32 or 96
    constexpr int FSTRIDE = KPAD + 4;           // bf16 units; keeps rows 8B-aligned

    constexpr int V4_PER_ROW  = FDIM / 4;        // float4 chunks per feat row (4|16)
    constexpr int ROWS_PER_IT = 32 / V4_PER_ROW; // edges covered per iteration (8|2)
    constexpr int GITER       = 16 / ROWS_PER_IT;// gather iterations (2|8)

    __shared__ __attribute__((aligned(16))) __bf16 lds_feat[8 * 16 * FSTRIDE];
    __shared__ __attribute__((aligned(16))) __bf16 lds_hid [8 * 16 * HID_STRIDE];

    const int lane = threadIdx.x & 31;
    const int wave = __builtin_amdgcn_readfirstlane((int)(threadIdx.x >> 5));
    __bf16* fT = &lds_feat[wave * 16 * FSTRIDE];
    __bf16* hT = &lds_hid [wave * 16 * HID_STRIDE];

    // Zero the whole feature tile once: pad columns stay zero forever,
    // real columns are overwritten every tile.
    for (int idx = lane; idx < 16 * FSTRIDE; idx += 32) fT[idx] = (__bf16)0.0f;

    // Hoist weight (B) fragments: invariant across the grid-stride loop.
    v16bf B1[NCHUNK * 2];
#pragma unroll
    for (int kc = 0; kc < NCHUNK; ++kc)
#pragma unroll
        for (int nb = 0; nb < 2; ++nb)
            B1[kc * 2 + nb] = load_B1_perm<FDIM>(W1, kc, nb, lane);
    v16bf B2[4];
#pragma unroll
    for (int nb = 0; nb < 4; ++nb)
        B2[nb] = load_B_frag(W2, MLP_H, H_DIM, 0, nb, lane);

    const float bias1a = b1[lane & 15];
    const float bias1b = b1[16 + (lane & 15)];
    float bias2[4];
#pragma unroll
    for (int nb = 0; nb < 4; ++nb) bias2[nb] = b2[nb * 16 + (lane & 15)];

    const int tiles  = NE / 16;
    const int nWaves = gridDim.x * (blockDim.x >> 5);
    const int n  = lane & 15;   // C/D column within 16-block
    const int h2 = lane >> 4;   // lane half

    for (int t = blockIdx.x * (blockDim.x >> 5) + wave; t < tiles; t += nWaves) {
        const int base = t * 16;

        // Per-lane metadata: lane L holds edge (base + (L&15)).
        const int eL = base + (lane & 15);
        const int sv = srcA[eL];
        const int dv = dstA[eL];
        const float xv = disA[eL];

        // dst indices into SGPRs for the scatter stage.
        int dsts[16];
#pragma unroll
        for (int ei = 0; ei < 16; ++ei)
            dsts[ei] = __builtin_amdgcn_readlane(dv, ei);

        // --- feat block: coalesced float4 loads -> 4x bf16 LDS stores -------
#pragma unroll
        for (int it = 0; it < GITER; ++it) {
            const int ei = it * ROWS_PER_IT + lane / V4_PER_ROW;
            const int q  = lane % V4_PER_ROW;           // float4 index in row
            const int s  = __shfl(sv, ei, 32);
            const float4 f4 = *(const float4*)(feat + (size_t)s * FDIM + q * 4);
            v4bf w; w[0] = (__bf16)f4.x; w[1] = (__bf16)f4.y;
                    w[2] = (__bf16)f4.z; w[3] = (__bf16)f4.w;
            *(v4bf*)(fT + ei * FSTRIDE + q * 4) = w;
        }

        // --- meta block: cols FDIM..FDIM+4 = [posSrc, posS[dst], dis] -------
        // Lane already holds its own edge's src/dst/dis; halves split columns.
#pragma unroll
        for (int j = 0; j < 3; ++j) {
            const int c = 2 * j + h2;                   // 0..5
            if (c < 5) {
                float v;
                if (c < 2)      v = posSrc[sv * 2 + c];
                else if (c < 4) v = posS[dv * 2 + (c - 2)];
                else            v = xv;
                fT[(lane & 15) * FSTRIDE + FDIM + c] = (__bf16)v;
            }
        }
        asm volatile("s_wait_dscnt 0" ::: "memory");  // wave-local store->load

        // --- GEMM1: [16 x KPAD] @ [KPAD x 32] (bias preloaded into C) -------
        v8f c0, c1;
#pragma unroll
        for (int r = 0; r < 8; ++r) { c0[r] = bias1a; c1[r] = bias1b; }
#pragma unroll
        for (int kc = 0; kc < NCHUNK; ++kc) {
            const v16bf a = load_A_lds(fT, FSTRIDE, kc, lane);
            c0 = __builtin_amdgcn_wmma_f32_16x16x32_bf16(false, a, false, B1[kc * 2 + 0],
                                                         (short)0, c0, false, false);
            c1 = __builtin_amdgcn_wmma_f32_16x16x32_bf16(false, a, false, B1[kc * 2 + 1],
                                                         (short)0, c1, false, false);
        }

        // --- LeakyReLU; stage hidden 16x32 tile (C-layout -> row-major) -----
        const int mb = h2 ? 8 : 0;
#pragma unroll
        for (int r = 0; r < 8; ++r) {
            float v0 = c0[r]; v0 = v0 > 0.0f ? v0 : 0.01f * v0;
            float v1 = c1[r]; v1 = v1 > 0.0f ? v1 : 0.01f * v1;
            hT[(mb + r) * HID_STRIDE + n]      = (__bf16)v0;
            hT[(mb + r) * HID_STRIDE + 16 + n] = (__bf16)v1;
        }
        asm volatile("s_wait_dscnt 0" ::: "memory");

        const v16bf a2 = load_A_lds(hT, HID_STRIDE, 0, lane);

        // --- GEMM2: [16 x 32] @ [32 x 64] -> tanh -> scatter-add ------------
        v8f co[4];
#pragma unroll
        for (int nb = 0; nb < 4; ++nb) {
#pragma unroll
            for (int r = 0; r < 8; ++r) co[nb][r] = bias2[nb];
            co[nb] = __builtin_amdgcn_wmma_f32_16x16x32_bf16(false, a2, false, B2[nb],
                                                             (short)0, co[nb], false, false);
        }

#pragma unroll
        for (int r = 0; r < 8; ++r) {
            const int dr = h2 ? dsts[8 + r] : dsts[r];
            float* rowp = outSum + (size_t)dr * H_DIM;
#pragma unroll
            for (int nb = 0; nb < 4; ++nb)
                atomicAdd(rowp + nb * 16 + n, tanhf(co[nb][r]));
        }
    }
}

// ---------------------------------------------------------------------------
// Node-update kernel: one wave == 16 states, 194->32->64 MLP, direct store.
// (3% of total FLOPs; rows are already contiguous per lane.)
// ---------------------------------------------------------------------------
static __device__ inline float node_feature(int k, int s,
                                            const float* __restrict__ posS,
                                            const float* __restrict__ h,
                                            const float* __restrict__ su,
                                            const float* __restrict__ sh) {
    if (k < 2)   return posS[s * 2 + k];
    if (k < 66)  return h[s * 64 + (k - 2)];
    if (k < 130) return su[s * 64 + (k - 66)];
    if (k < 194) return sh[s * 64 + (k - 130)];
    return 0.0f;
}

__global__ __launch_bounds__(256)
void node_update_kernel(const float* __restrict__ posS, const float* __restrict__ h,
                        const float* __restrict__ sumU, const float* __restrict__ sumH,
                        const float* __restrict__ W1, const float* __restrict__ b1,
                        const float* __restrict__ W2, const float* __restrict__ b2,
                        float* __restrict__ out) {
    constexpr int KDIM   = 2 + 3 * H_DIM;        // 194
    constexpr int NCHUNK = (KDIM + 31) / 32;     // 7

    __shared__ __attribute__((aligned(16))) __bf16 lds_hid[8 * 16 * HID_STRIDE];
    const int lane = threadIdx.x & 31;
    const int wave = __builtin_amdgcn_readfirstlane((int)(threadIdx.x >> 5));
    __bf16* hT = &lds_hid[wave * 16 * HID_STRIDE];

    v16bf B2[4];
#pragma unroll
    for (int nb = 0; nb < 4; ++nb)
        B2[nb] = load_B_frag(W2, MLP_H, H_DIM, 0, nb, lane);

    const float bias1a = b1[lane & 15];
    const float bias1b = b1[16 + (lane & 15)];
    float bias2[4];
#pragma unroll
    for (int nb = 0; nb < 4; ++nb) bias2[nb] = b2[nb * 16 + (lane & 15)];

    const int tiles  = N_S / 16;
    const int nWaves = gridDim.x * (blockDim.x >> 5);
    const int n  = lane & 15;
    const int kb = (lane & 16) ? 8 : 0;
    const int mb = (lane & 16) ? 8 : 0;

    for (int t = blockIdx.x * (blockDim.x >> 5) + wave; t < tiles; t += nWaves) {
        const int base = t * 16;
        const int s    = base + (lane & 15);

        v8f c0, c1;
#pragma unroll
        for (int r = 0; r < 8; ++r) { c0[r] = bias1a; c1[r] = bias1b; }

#pragma unroll
        for (int kc = 0; kc < NCHUNK; ++kc) {
            v16bf a;
#pragma unroll
            for (int j = 0; j < 8; ++j) {
                const int k0 = kc * 32 + kb + j;
                const int k1 = kc * 32 + kb + 16 + j;
                a[j]     = (__bf16)node_feature(k0, s, posS, h, sumU, sumH);
                a[8 + j] = (__bf16)node_feature(k1, s, posS, h, sumU, sumH);
            }
            const v16bf b1f0 = load_B_frag(W1, KDIM, MLP_H, kc, 0, lane);
            const v16bf b1f1 = load_B_frag(W1, KDIM, MLP_H, kc, 1, lane);
            c0 = __builtin_amdgcn_wmma_f32_16x16x32_bf16(false, a, false, b1f0,
                                                         (short)0, c0, false, false);
            c1 = __builtin_amdgcn_wmma_f32_16x16x32_bf16(false, a, false, b1f1,
                                                         (short)0, c1, false, false);
        }

#pragma unroll
        for (int r = 0; r < 8; ++r) {
            float v0 = c0[r]; v0 = v0 > 0.0f ? v0 : 0.01f * v0;
            float v1 = c1[r]; v1 = v1 > 0.0f ? v1 : 0.01f * v1;
            hT[(mb + r) * HID_STRIDE + n]      = (__bf16)v0;
            hT[(mb + r) * HID_STRIDE + 16 + n] = (__bf16)v1;
        }
        asm volatile("s_wait_dscnt 0" ::: "memory");

        const v16bf a2 = load_A_lds(hT, HID_STRIDE, 0, lane);

        v8f co[4];
#pragma unroll
        for (int nb = 0; nb < 4; ++nb) {
#pragma unroll
            for (int r = 0; r < 8; ++r) co[nb][r] = bias2[nb];
            co[nb] = __builtin_amdgcn_wmma_f32_16x16x32_bf16(false, a2, false, B2[nb],
                                                             (short)0, co[nb], false, false);
        }

#pragma unroll
        for (int r = 0; r < 8; ++r) {
            float* rowp = out + (size_t)(base + mb + r) * H_DIM;
#pragma unroll
            for (int nb = 0; nb < 4; ++nb)
                rowp[nb * 16 + n] = tanhf(co[nb][r]);
        }
    }
}

// ---------------------------------------------------------------------------
extern "C" void kernel_launch(void* const* d_in, const int* in_sizes, int n_in,
                              void* d_out, int out_size, void* d_ws, size_t ws_size,
                              hipStream_t stream) {
    const float* h       = (const float*)d_in[0];
    const float* u       = (const float*)d_in[1];
    const float* pos_s   = (const float*)d_in[2];
    const float* pos_a   = (const float*)d_in[3];
    const float* dis_a2s = (const float*)d_in[4];
    const float* dis_s2s = (const float*)d_in[5];
    const int*   a2s_src = (const int*)d_in[6];
    const int*   a2s_dst = (const int*)d_in[7];
    const int*   s2s_src = (const int*)d_in[8];
    const int*   s2s_dst = (const int*)d_in[9];
    const float* a2s_W1  = (const float*)d_in[10];
    const float* a2s_b1  = (const float*)d_in[11];
    const float* a2s_W2  = (const float*)d_in[12];
    const float* a2s_b2  = (const float*)d_in[13];
    const float* s2s_W1  = (const float*)d_in[14];
    const float* s2s_b1  = (const float*)d_in[15];
    const float* s2s_W2  = (const float*)d_in[16];
    const float* s2s_b2  = (const float*)d_in[17];
    const float* upd_W1  = (const float*)d_in[18];
    const float* upd_b1  = (const float*)d_in[19];
    const float* upd_W2  = (const float*)d_in[20];
    const float* upd_b2  = (const float*)d_in[21];

    float* sum_u = (float*)d_ws;
    float* sum_h = sum_u + (size_t)N_S * H_DIM;

    hipMemsetAsync(d_ws, 0, 2 * (size_t)N_S * H_DIM * sizeof(float), stream);

    dim3 blk(256);
    // 1024 blocks * 8 waves = 8192 waves over 100000 16-edge tiles.
    edge_msg_kernel<U_DIM><<<1024, blk, 0, stream>>>(
        a2s_src, a2s_dst, dis_a2s, pos_a, pos_s, u,
        a2s_W1, a2s_b1, a2s_W2, a2s_b2, sum_u);
    edge_msg_kernel<H_DIM><<<1024, blk, 0, stream>>>(
        s2s_src, s2s_dst, dis_s2s, pos_s, pos_s, h,
        s2s_W1, s2s_b1, s2s_W2, s2s_b2, sum_h);
    // 3136 waves over 3125 16-state tiles.
    node_update_kernel<<<392, blk, 0, stream>>>(
        pos_s, h, sum_u, sum_h,
        upd_W1, upd_b1, upd_W2, upd_b2, (float*)d_out);
}
